// MTAAttention_54803782697222
// MI455X (gfx1250) — compile-verified, matrix operand order checked by reading, not run on hardware
//
#include <hip/hip_runtime.h>

#define Bsz 8
#define Nn  1024
#define Cc  384
#define Hh  6
#define Dd  64
#define SCALE 0.125f
#define EPS 1e-5f

typedef __attribute__((ext_vector_type(16))) __bf16 v16bf;
typedef __attribute__((ext_vector_type(8)))  __bf16 v8bf;
typedef __attribute__((ext_vector_type(8)))  float  v8f;

union Frag { v16bf v; v8bf h[2]; };

__device__ inline __bf16 f2bf(float f) {
    unsigned u = __builtin_bit_cast(unsigned, f);
    unsigned r = u + 0x7FFFu + ((u >> 16) & 1u);   // RNE
    unsigned short s = (unsigned short)(r >> 16);
    return __builtin_bit_cast(__bf16, s);
}
__device__ inline float bf2f(__bf16 x) {
    unsigned short s = __builtin_bit_cast(unsigned short, x);
    unsigned u = ((unsigned)s) << 16;
    return __builtin_bit_cast(float, u);
}

__device__ inline v8f wmma_bf16(const Frag& a, const Frag& b, v8f c) {
    return __builtin_amdgcn_wmma_f32_16x16x32_bf16(false, a.v, false, b.v,
                                                   (short)0, c, false, false);
}

// 2x2 register-blocked K-loop: wave computes a 32x32 fp32 block of A*B^T.
// A[M,K] row-major (lda), B given as [Nout,K] row-major (ldb).
// ISA 16-bit fragment layout: lanes 0-15 hold K-chunks {0..7,16..23},
// lanes 16-31 hold {8..15,24..31}; per k-step: 2 A-frags + 2 B-frags -> 4 WMMAs.
__device__ inline void wmma_kloop22(const __bf16* __restrict__ A, int lda,
                                    const __bf16* __restrict__ B, int ldb,
                                    int r0, int c0, int K, v8f acc[2][2], int lane) {
    const int lr = lane & 15;
    const int kh = (lane >> 4) << 3;   // 0 or 8
    const __bf16* pa0 = A + (size_t)(r0 + lr) * lda + kh;
    const __bf16* pa1 = pa0 + (size_t)16 * lda;
    const __bf16* pb0 = B + (size_t)(c0 + lr) * ldb + kh;
    const __bf16* pb1 = pb0 + (size_t)16 * ldb;
    for (int kb = 0; kb < K; kb += 32) {
        Frag a0, a1, b0, b1;
        a0.h[0] = *(const v8bf*)(pa0 + kb); a0.h[1] = *(const v8bf*)(pa0 + kb + 16);
        a1.h[0] = *(const v8bf*)(pa1 + kb); a1.h[1] = *(const v8bf*)(pa1 + kb + 16);
        b0.h[0] = *(const v8bf*)(pb0 + kb); b0.h[1] = *(const v8bf*)(pb0 + kb + 16);
        b1.h[0] = *(const v8bf*)(pb1 + kb); b1.h[1] = *(const v8bf*)(pb1 + kb + 16);
        acc[0][0] = wmma_bf16(a0, b0, acc[0][0]);
        acc[0][1] = wmma_bf16(a0, b1, acc[0][1]);
        acc[1][0] = wmma_bf16(a1, b0, acc[1][0]);
        acc[1][1] = wmma_bf16(a1, b1, acc[1][1]);
    }
}

// ---------------- stage 0: fp32 -> bf16 convert ----------------
__global__ void cvt_kernel(const float* __restrict__ src, __bf16* __restrict__ dst, int n) {
    int i = blockIdx.x * 256 + threadIdx.x;
    if (i < n) dst[i] = f2bf(src[i]);
}

// ---------------- stage 1: QKV projection ----------------
// C[8192,1152] = xb[8192,384] * wqkvb[1152,384]^T, scattered into q/k/v [B,H,N,D] bf16
__global__ __launch_bounds__(256) void qkv_kernel(const __bf16* __restrict__ xb,
                                                  const __bf16* __restrict__ wq,
                                                  __bf16* __restrict__ qb,
                                                  __bf16* __restrict__ kb,
                                                  __bf16* __restrict__ vb) {
    int lane = threadIdx.x & 31, wave = threadIdx.x >> 5;
    int r0 = blockIdx.x * 64 + (wave >> 2) * 32;
    int c0 = blockIdx.y * 128 + (wave & 3) * 32;
    v8f acc[2][2] = {};
    wmma_kloop22(xb, Cc, wq, Cc, r0, c0, Cc, acc, lane);
#pragma unroll
    for (int j = 0; j < 2; j++) {
        int col = c0 + j * 16 + (lane & 15);
        int t = col / Cc; int rem = col % Cc; int h = rem >> 6; int d = rem & 63;
        __bf16* dst = (t == 0) ? qb : (t == 1) ? kb : vb;
        float sc = (t == 0) ? SCALE : 1.0f;
#pragma unroll
        for (int i = 0; i < 2; i++) {
#pragma unroll
            for (int r = 0; r < 8; r++) {
                int m = r0 + i * 16 + r + ((lane >> 4) << 3);
                int b = m >> 10, n = m & 1023;
                dst[(((size_t)(b * Hh + h) * Nn + n) << 6) + d] = f2bf(acc[i][j][r] * sc);
            }
        }
    }
}

// ---------------- stage 2: logits = q @ k^T (SCALE folded into q) ----------------
__global__ __launch_bounds__(256) void logits_kernel(const __bf16* __restrict__ qb,
                                                     const __bf16* __restrict__ kb,
                                                     float* __restrict__ logits) {
    int lane = threadIdx.x & 31, wave = threadIdx.x >> 5;
    int bh = blockIdx.z;
    int r0 = blockIdx.x * 64 + (wave >> 2) * 32;
    int c0 = blockIdx.y * 128 + (wave & 3) * 32;
    const __bf16* A = qb + (size_t)bh * Nn * Dd;
    const __bf16* B = kb + (size_t)bh * Nn * Dd;
    v8f acc[2][2] = {};
    wmma_kloop22(A, Dd, B, Dd, r0, c0, Dd, acc, lane);
    float* out = logits + (size_t)bh * Nn * Nn;
#pragma unroll
    for (int i = 0; i < 2; i++) {
#pragma unroll
        for (int j = 0; j < 2; j++) {
            int col = c0 + j * 16 + (lane & 15);
#pragma unroll
            for (int r = 0; r < 8; r++) {
                int row = r0 + i * 16 + r + ((lane >> 4) << 3);
                out[(size_t)row * Nn + col] = acc[i][j][r];
            }
        }
    }
}

// ---------------- stage 3: fused conv3x3 + softmax + head-mix + GN stats ----------------
__device__ inline float wave_red(float v, bool is_max) {
#pragma unroll
    for (int off = 16; off; off >>= 1) {
        float o = __shfl_xor(v, off, 32);
        v = is_max ? fmaxf(v, o) : v + o;
    }
    return v;
}
__device__ inline float block_red(float v, bool is_max, float* red) {
    int lane = threadIdx.x & 31, wid = threadIdx.x >> 5;
    v = wave_red(v, is_max);
    __syncthreads();
    if (lane == 0) red[wid] = v;
    __syncthreads();
    float r = red[0];
#pragma unroll
    for (int j = 1; j < 8; j++) r = is_max ? fmaxf(r, red[j]) : r + red[j];
    return r;
}

__global__ __launch_bounds__(256) void csm_kernel(const float* __restrict__ logits,
                                                  const float* __restrict__ w_kq,
                                                  const float* __restrict__ b_kq,
                                                  const float* __restrict__ w_head,
                                                  const float* __restrict__ b_head,
                                                  __bf16* __restrict__ mixed,
                                                  float* __restrict__ stats) {
    __shared__ float sm[Hh][Nn];
    __shared__ float red[8];
    int bn = blockIdx.x;
    int b = bn >> 10, n = bn & 1023;
    int tid = threadIdx.x;
    int m0 = tid * 4;

    for (int h = 0; h < Hh; h++) {
        float w[9];
#pragma unroll
        for (int i = 0; i < 9; i++) w[i] = w_kq[h * 9 + i];
        float bias = b_kq[h];
        const float* Lh = logits + (size_t)(b * Hh + h) * Nn * Nn;
        float row[3][6];
#pragma unroll
        for (int rr = 0; rr < 3; rr++) {
            int nn = n - 1 + rr;
#pragma unroll
            for (int c = 0; c < 6; c++) {
                int mm = m0 - 1 + c;
                float val = 0.f;
                if (nn >= 0 && nn < Nn && mm >= 0 && mm < Nn)
                    val = Lh[(size_t)nn * Nn + mm];
                row[rr][c] = val;
            }
        }
        float vals[4];
        float lmax = -3.4e38f;
#pragma unroll
        for (int c = 0; c < 4; c++) {
            float s = bias;
#pragma unroll
            for (int i = 0; i < 3; i++)
#pragma unroll
                for (int j = 0; j < 3; j++)
                    s += w[i * 3 + j] * row[i][c + j];
            vals[c] = s;
            lmax = fmaxf(lmax, s);
        }
        float rmax = block_red(lmax, true, red);
        float lsum = 0.f;
#pragma unroll
        for (int c = 0; c < 4; c++) { vals[c] = __expf(vals[c] - rmax); lsum += vals[c]; }
        float rsum = block_red(lsum, false, red);
        float inv = 1.0f / rsum;
#pragma unroll
        for (int c = 0; c < 4; c++) sm[h][m0 + c] = vals[c] * inv;
    }
    __syncthreads();

    float wmix[Hh * Hh], bmix[Hh];
#pragma unroll
    for (int i = 0; i < Hh * Hh; i++) wmix[i] = w_head[i];
#pragma unroll
    for (int o = 0; o < Hh; o++) bmix[o] = b_head[o];

    float lsum[Hh] = {}, lsq[Hh] = {};
#pragma unroll
    for (int c = 0; c < 4; c++) {
        int m = m0 + c;
        float s[Hh];
#pragma unroll
        for (int h = 0; h < Hh; h++) s[h] = sm[h][m];
#pragma unroll
        for (int o = 0; o < Hh; o++) {
            float mv = bmix[o];
#pragma unroll
            for (int h = 0; h < Hh; h++) mv += wmix[o * Hh + h] * s[h];
            mixed[((size_t)(b * Hh + o) * Nn + n) * Nn + m] = f2bf(mv);
            lsum[o] += mv;
            lsq[o] += mv * mv;
        }
    }
    for (int o = 0; o < Hh; o++) {
        float rs = block_red(lsum[o], false, red);
        float rq = block_red(lsq[o], false, red);
        if (tid == 0) {
            atomicAdd(&stats[(b * Hh + o) * 2 + 0], rs);
            atomicAdd(&stats[(b * Hh + o) * 2 + 1], rq);
        }
    }
}

// ---------------- stage 4a: v colsums + transpose ----------------
__global__ void vprep_kernel(const __bf16* __restrict__ vb, __bf16* __restrict__ vT,
                             float* __restrict__ sumv) {
    int bh = blockIdx.x, d = threadIdx.x;
    const __bf16* vsrc = vb + (size_t)bh * Nn * Dd;
    __bf16* vt = vT + (size_t)bh * Dd * Nn;
    float acc = 0.f;
    for (int m = 0; m < Nn; m++) {
        __bf16 x = vsrc[(size_t)m * Dd + d];
        acc += bf2f(x);
        vt[(size_t)d * Nn + m] = x;
    }
    sumv[bh * Dd + d] = acc;
}

// ---------------- stage 4b: GroupNorm affine finalize ----------------
__global__ void zero_kernel(float* p, int n) {
    int i = blockIdx.x * 256 + threadIdx.x;
    if (i < n) p[i] = 0.f;
}
__global__ void gnfin_kernel(const float* __restrict__ stats,
                             const float* __restrict__ gn_w,
                             const float* __restrict__ gn_b,
                             float* __restrict__ ab) {
    int t = threadIdx.x;
    if (t >= Bsz * Hh) return;
    float cnt = (float)Nn * (float)Nn;
    float mean = stats[t * 2 + 0] / cnt;
    float var  = stats[t * 2 + 1] / cnt - mean * mean;
    float rs = rsqrtf(var + EPS);
    int h = t % Hh;
    float alpha = gn_w[h] * rs;
    float beta  = gn_b[h] - mean * alpha;
    ab[t * 2 + 0] = alpha;
    ab[t * 2 + 1] = beta;
}

// ---------------- stage 5: ctx = (alpha*mixed + beta) @ v  (beta folded via colsum) ----------------
__global__ __launch_bounds__(256) void av_kernel(const __bf16* __restrict__ mixed,
                                                 const __bf16* __restrict__ vT,
                                                 const float* __restrict__ ab,
                                                 const float* __restrict__ sumv,
                                                 __bf16* __restrict__ ctxb) {
    int lane = threadIdx.x & 31, wave = threadIdx.x >> 5;
    int bh = blockIdx.z;
    int r0 = blockIdx.x * 128 + (wave >> 1) * 32;
    int c0 = (wave & 1) * 32;                 // Nout = 64: two waves cover it
    const __bf16* A = mixed + (size_t)bh * Nn * Nn;
    const __bf16* B = vT + (size_t)bh * Dd * Nn;
    v8f acc[2][2] = {};
    wmma_kloop22(A, Nn, B, Nn, r0, c0, Nn, acc, lane);
    float alpha = ab[bh * 2 + 0], beta = ab[bh * 2 + 1];
    int b = bh / Hh, h = bh % Hh;
#pragma unroll
    for (int j = 0; j < 2; j++) {
        int d = c0 + j * 16 + (lane & 15);
        float corr = beta * sumv[bh * Dd + d];
#pragma unroll
        for (int i = 0; i < 2; i++) {
#pragma unroll
            for (int r = 0; r < 8; r++) {
                int n = r0 + i * 16 + r + ((lane >> 4) << 3);
                ctxb[((size_t)(b * Nn + n)) * Cc + h * Dd + d] = f2bf(alpha * acc[i][j][r] + corr);
            }
        }
    }
}

// ---------------- stage 6: out = ctx @ w_proj^T + b_proj ----------------
__global__ __launch_bounds__(256) void proj_kernel(const __bf16* __restrict__ ctxb,
                                                   const __bf16* __restrict__ wp,
                                                   const float* __restrict__ b_proj,
                                                   float* __restrict__ out) {
    int lane = threadIdx.x & 31, wave = threadIdx.x >> 5;
    int r0 = blockIdx.x * 64 + (wave >> 2) * 32;
    int c0 = blockIdx.y * 128 + (wave & 3) * 32;
    v8f acc[2][2] = {};
    wmma_kloop22(ctxb, Cc, wp, Cc, r0, c0, Cc, acc, lane);
#pragma unroll
    for (int j = 0; j < 2; j++) {
        int col = c0 + j * 16 + (lane & 15);
        float bias = b_proj[col];
#pragma unroll
        for (int i = 0; i < 2; i++) {
#pragma unroll
            for (int r = 0; r < 8; r++) {
                int m = r0 + i * 16 + r + ((lane >> 4) << 3);
                out[(size_t)m * Cc + col] = acc[i][j][r] + bias;
            }
        }
    }
}

extern "C" void kernel_launch(void* const* d_in, const int* in_sizes, int n_in,
                              void* d_out, int out_size, void* d_ws, size_t ws_size,
                              hipStream_t stream) {
    const float* x      = (const float*)d_in[0];
    const float* w_qkv  = (const float*)d_in[1];
    const float* w_proj = (const float*)d_in[2];
    const float* b_proj = (const float*)d_in[3];
    const float* w_kq   = (const float*)d_in[4];
    const float* b_kq   = (const float*)d_in[5];
    const float* w_head = (const float*)d_in[6];
    const float* b_head = (const float*)d_in[7];
    const float* gn_w   = (const float*)d_in[8];
    const float* gn_b   = (const float*)d_in[9];
    float* out = (float*)d_out;

    char* ws = (char*)d_ws;
    size_t off = 0;
    auto alloc = [&](size_t bytes) {
        size_t o = off;
        off = (off + bytes + 255) & ~(size_t)255;
        return o;
    };
    const size_t BH = (size_t)Bsz * Hh;
    float*  logits = (float*) (ws + alloc(BH * Nn * Nn * 4));
    __bf16* mixed  = (__bf16*)(ws + alloc(BH * Nn * Nn * 2));
    __bf16* xb     = (__bf16*)(ws + alloc((size_t)Bsz * Nn * Cc * 2));
    __bf16* wqkvb  = (__bf16*)(ws + alloc((size_t)3 * Cc * Cc * 2));
    __bf16* wprojb = (__bf16*)(ws + alloc((size_t)Cc * Cc * 2));
    __bf16* qb     = (__bf16*)(ws + alloc(BH * Nn * Dd * 2));
    __bf16* kb     = (__bf16*)(ws + alloc(BH * Nn * Dd * 2));
    __bf16* vb     = (__bf16*)(ws + alloc(BH * Nn * Dd * 2));
    __bf16* vT     = (__bf16*)(ws + alloc(BH * Nn * Dd * 2));
    __bf16* ctxb   = (__bf16*)(ws + alloc((size_t)Bsz * Nn * Cc * 2));
    float*  stats  = (float*) (ws + alloc(BH * 2 * 4));
    float*  ab     = (float*) (ws + alloc(BH * 2 * 4));
    float*  sumv   = (float*) (ws + alloc(BH * Dd * 4));
    (void)ws_size; (void)in_sizes; (void)n_in; (void)out_size;

    // stage 0: converts
    {
        int n0 = Bsz * Nn * Cc;
        cvt_kernel<<<(n0 + 255) / 256, 256, 0, stream>>>(x, xb, n0);
        int n1 = 3 * Cc * Cc;
        cvt_kernel<<<(n1 + 255) / 256, 256, 0, stream>>>(w_qkv, wqkvb, n1);
        int n2 = Cc * Cc;
        cvt_kernel<<<(n2 + 255) / 256, 256, 0, stream>>>(w_proj, wprojb, n2);
    }
    // stage 1: QKV projection (M=8192, N=1152, K=384); block tile 64x128
    qkv_kernel<<<dim3(8192 / 64, 1152 / 128), 256, 0, stream>>>(xb, wqkvb, qb, kb, vb);
    // stage 2: logits (per b,h: 1024x1024x64); block tile 64x128
    logits_kernel<<<dim3(Nn / 64, Nn / 128, (unsigned)BH), 256, 0, stream>>>(qb, kb, logits);
    // stage 3: fused conv+softmax+mix+stats
    zero_kernel<<<1, 256, 0, stream>>>(stats, (int)(BH * 2));
    csm_kernel<<<Bsz * Nn, 256, 0, stream>>>(logits, w_kq, b_kq, w_head, b_head, mixed, stats);
    // stage 4: v prep + GN affine
    vprep_kernel<<<(unsigned)BH, Dd, 0, stream>>>(vb, vT, sumv);
    gnfin_kernel<<<1, 64, 0, stream>>>(stats, gn_w, gn_b, ab);
    // stage 5: attn @ v with folded GroupNorm (per b,h: 1024x64x1024); block tile 128x64
    av_kernel<<<dim3(Nn / 128, 1, (unsigned)BH), 256, 0, stream>>>(mixed, vT, ab, sumv, ctxb);
    // stage 6: output projection (M=8192, N=384, K=384); block tile 64x128
    proj_kernel<<<dim3(8192 / 64, Cc / 128), 256, 0, stream>>>(ctxb, wprojb, b_proj, out);
}
